// SpectralGNNEncoder_59184649339354
// MI455X (gfx1250) — compile-verified, matrix operand order checked by reading, not run on hardware
//
#include <hip/hip_runtime.h>

// ---------------------------------------------------------------------------
// CDNA5 (gfx1250) GCN-VAE encoder.
// GEMMs: v_wmma_f32_16x16x32_bf16, weights staged via global_load_async_to_lds.
// Edge aggregation: fp32 HW atomics (L2-resident working set).
// ---------------------------------------------------------------------------

typedef __attribute__((ext_vector_type(16))) __bf16 v16bf;
typedef __attribute__((ext_vector_type(8)))  float  v8f;

__device__ __forceinline__ unsigned short f2bf(float f) {
    union { float f; unsigned u; } v; v.f = f;
    unsigned u = v.u;
    // round-to-nearest-even f32 -> bf16
    return (unsigned short)((u + 0x7FFFu + ((u >> 16) & 1u)) >> 16);
}

__global__ void zero_f32(float* __restrict__ p, int n) {
    int i = blockIdx.x * blockDim.x + threadIdx.x;
    int stride = gridDim.x * blockDim.x;
    for (; i < n; i += stride) p[i] = 0.0f;
}

// Wt[n*128 + k] = bf16(W[k*128 + n])  (W is 128x128 row-major, k = input dim)
__global__ void prep_weights(const float* __restrict__ W, unsigned short* __restrict__ Wt) {
    int i = blockIdx.x * blockDim.x + threadIdx.x;
    int stride = gridDim.x * blockDim.x;
    for (; i < 128 * 128; i += stride) {
        int n = i >> 7, k = i & 127;
        Wt[n * 128 + k] = f2bf(W[k * 128 + n]);
    }
}

// weighted in-degree at dst (self-loop "+1" folded into dinv_kernel)
__global__ void deg_kernel(const int* __restrict__ dst, const float* __restrict__ w,
                           float* __restrict__ deg, int E) {
    int e = blockIdx.x * blockDim.x + threadIdx.x;
    if (e < E) unsafeAtomicAdd(&deg[dst[e]], w[e]);
}

__global__ void dinv_kernel(const float* __restrict__ deg, float* __restrict__ dinv, int N) {
    int i = blockIdx.x * blockDim.x + threadIdx.x;
    if (i < N) dinv[i] = rsqrtf(deg[i] + 1.0f);   // deg+1 > 0 always (self-loop weight 1)
}

#define TILES_PER_BLOCK 4

// Y[N,128] = X[N,128] @ W[128,128]   (Wt is bf16, pre-transposed [n][k])
// 256 threads = 8 waves; each block computes 4 consecutive 16-row tiles,
// reusing the LDS-resident 32KB bf16 weight matrix (staged asynchronously).
__global__ __launch_bounds__(256)
void gemm_bf16_wmma(const float* __restrict__ X, const unsigned short* __restrict__ Wt,
                    float* __restrict__ Y, int N) {
    __shared__ unsigned short sW[128 * 128];  // 32 KB bf16 weights ([n][k])
    __shared__ unsigned short sX[16 * 128];   // 4 KB bf16 x-tile

    const int tid = threadIdx.x;

    // ---- async-stage Wt -> LDS (2048 x b128, ASYNCcnt-tracked) ----
    {
        const unsigned swBase = (unsigned)(unsigned long long)(&sW[0]); // LDS byte offset
        const unsigned long long gBase = (unsigned long long)Wt;
#pragma unroll
        for (int j = 0; j < 8; ++j) {
            unsigned      ldsa = swBase + (unsigned)(tid + 256 * j) * 16u;
            unsigned long long ga = gBase + (unsigned long long)(tid + 256 * j) * 16ull;
            asm volatile("global_load_async_to_lds_b128 %0, %1, off"
                         :: "v"(ldsa), "v"(ga) : "memory");
        }
    }

    const int lane  = tid & 31;
    const int wave  = tid >> 5;
    const int m     = lane & 15;                  // A row within tile
    const int ncol  = wave * 16 + (lane & 15);    // output column
    const int kb0   = (lane >> 4) * 8;            // K sub-base per half-wave
    const int rbase = (lane >> 4) * 8;            // C/D row base

    for (int t = 0; t < TILES_PER_BLOCK; ++t) {
        const int row0 = (blockIdx.x * TILES_PER_BLOCK + t) * 16;
        if (row0 >= N) break;
        const bool full = (row0 + 16 <= N);

        if (t) __syncthreads();                   // protect sX rewrite

        // prefetch next x-tile into caches (global_prefetch_b8)
        if (row0 + 16 < N)
            __builtin_prefetch(&X[(size_t)(row0 + 16) * 128 + tid * 8], 0, 0);

        // stage x tile (16x128 f32 -> bf16), overlapped with async W copy (t==0)
#pragma unroll
        for (int j = 0; j < 8; ++j) {
            int idx = tid + 256 * j;              // 0..2047
            int r = idx >> 7, c = idx & 127;
            int gr = row0 + r;
            float v;
            if (full)           v = X[(size_t)gr * 128 + c];
            else if (gr < N)    v = X[(size_t)gr * 128 + c];
            else                v = 0.0f;
            sX[idx] = f2bf(v);
        }
        if (t == 0) asm volatile("s_wait_asynccnt 0x0" ::: "memory");
        __syncthreads();

        v8f acc = {};
        union Frag { v16bf v; uint4 q[2]; };
#pragma unroll
        for (int kt = 0; kt < 4; ++kt) {
            const int kb = kt * 32 + kb0;
            Frag a, b;
            // elements 0-7 -> K=kb..kb+7 ; elements 8-15 -> K=kb+16..kb+23
            a.q[0] = *(const uint4*)&sX[m * 128 + kb];
            a.q[1] = *(const uint4*)&sX[m * 128 + kb + 16];
            b.q[0] = *(const uint4*)&sW[ncol * 128 + kb];
            b.q[1] = *(const uint4*)&sW[ncol * 128 + kb + 16];
            acc = __builtin_amdgcn_wmma_f32_16x16x32_bf16(
                /*neg_a=*/false, a.v, /*neg_b=*/false, b.v,
                /*c_mod=*/(short)0, acc, /*reuse_a=*/false, /*reuse_b=*/false);
        }

        if (full) {
            // block-uniform fast path: no exec-mask juggling
            float* yb = Y + (size_t)(row0 + rbase) * 128 + ncol;
#pragma unroll
            for (int r = 0; r < 8; ++r) yb[(size_t)r * 128] = acc[r];
        } else {
#pragma unroll
            for (int r = 0; r < 8; ++r) {
                int gr = row0 + rbase + r;
                if (gr < N) Y[(size_t)gr * 128 + ncol] = acc[r];
            }
        }
    }
}

// one wave per edge: gather 128 floats of H[src], scale by dinv[s]*w*dinv[d],
// atomic scatter into Out[dst]. Both H and Out (51 MB each) are L2-resident.
__global__ __launch_bounds__(256)
void edge_scatter(const int* __restrict__ src, const int* __restrict__ dst,
                  const float* __restrict__ w, const float* __restrict__ dinv,
                  const float* __restrict__ H, float* __restrict__ Out, int E) {
    const int wave = threadIdx.x >> 5;
    const int lane = threadIdx.x & 31;
    const int e = blockIdx.x * 8 + wave;
    if (e >= E) return;
    const int s = src[e], d = dst[e];
    const float nrm = dinv[s] * w[e] * dinv[d];
    const float4 v = ((const float4*)(H + (size_t)s * 128))[lane];
    float* orow = Out + (size_t)d * 128 + lane * 4;
    unsafeAtomicAdd(&orow[0], v.x * nrm);
    unsafeAtomicAdd(&orow[1], v.y * nrm);
    unsafeAtomicAdd(&orow[2], v.z * nrm);
    unsafeAtomicAdd(&orow[3], v.w * nrm);
}

// Out[i,f] += dinv[i]^2 * H[i,f] + bias[f]; optional ReLU
__global__ void self_bias(const float* __restrict__ H, const float* __restrict__ dinv,
                          const float* __restrict__ bias, float* __restrict__ Out,
                          int N, int do_relu) {
    int i = blockIdx.x * blockDim.x + threadIdx.x;
    const int total = N * 128;
    const int stride = gridDim.x * blockDim.x;
    for (; i < total; i += stride) {
        const int node = i >> 7, f = i & 127;
        const float di = dinv[node];
        float v = Out[i] + di * di * H[i] + bias[f];
        Out[i] = do_relu ? fmaxf(v, 0.0f) : v;
    }
}

// column sums for mean pool (128 threads/block, one column per thread)
__global__ void col_sum(const float* __restrict__ H, float* __restrict__ gsum, int N) {
    const int f = threadIdx.x;
    float s = 0.0f;
    for (int r = blockIdx.x; r < N; r += gridDim.x) s += H[(size_t)r * 128 + f];
    unsafeAtomicAdd(&gsum[f], s);
}

// g = gsum/N ; out[0:64] = g@Wmu + bmu ; out[64:128] = g@Wlv + blv
__global__ void head_kernel(const float* __restrict__ gsum,
                            const float* __restrict__ Wmu, const float* __restrict__ bmu,
                            const float* __restrict__ Wlv, const float* __restrict__ blv,
                            float* __restrict__ out, int N) {
    const int j = threadIdx.x;          // 0..127
    const float inv = 1.0f / (float)N;
    const float* W = (j < 64) ? Wmu : Wlv;
    const float* b = (j < 64) ? bmu : blv;
    const int jj = j & 63;
    float s = 0.0f;
    for (int k = 0; k < 128; ++k) s += (gsum[k] * inv) * W[k * 64 + jj];
    out[j] = s + b[jj];
}

extern "C" void kernel_launch(void* const* d_in, const int* in_sizes, int n_in,
                              void* d_out, int out_size, void* d_ws, size_t ws_size,
                              hipStream_t stream) {
    (void)n_in; (void)out_size; (void)ws_size;
    const float* x   = (const float*)d_in[0];
    const int*   ei  = (const int*)d_in[1];
    const float* ew  = (const float*)d_in[2];
    const float* W1  = (const float*)d_in[3];
    const float* b1  = (const float*)d_in[4];
    const float* W2  = (const float*)d_in[5];
    const float* b2  = (const float*)d_in[6];
    const float* Wmu = (const float*)d_in[7];
    const float* bmu = (const float*)d_in[8];
    const float* Wlv = (const float*)d_in[9];
    const float* blv = (const float*)d_in[10];

    const int N = in_sizes[0] / 128;
    const int E = in_sizes[2];
    const int* src = ei;
    const int* dst = ei + E;

    // workspace layout (floats)
    float* ws   = (float*)d_ws;
    float* deg  = ws;                        // N
    float* dinv = deg + N;                   // N
    float* B0   = dinv + N;                  // N*128
    float* B1   = B0 + (size_t)N * 128;      // N*128
    float* gsum = B1 + (size_t)N * 128;      // 128
    unsigned short* W1t = (unsigned short*)(gsum + 128); // 128*128 bf16
    unsigned short* W2t = W1t + 128 * 128;               // 128*128 bf16

    const int NF = N * 128;
    const int gemmGrid = (N + 16 * TILES_PER_BLOCK - 1) / (16 * TILES_PER_BLOCK);
    const int edgeGrid = (E + 7) / 8;

    // degree / norm
    zero_f32<<<256, 256, 0, stream>>>(deg, N);
    prep_weights<<<64, 256, 0, stream>>>(W1, W1t);
    prep_weights<<<64, 256, 0, stream>>>(W2, W2t);
    deg_kernel<<<(E + 255) / 256, 256, 0, stream>>>(dst, ew, deg, E);
    dinv_kernel<<<(N + 255) / 256, 256, 0, stream>>>(deg, dinv, N);

    // layer 1: h_pre = x@W1 ; aggregate ; +self +b1 ; relu
    gemm_bf16_wmma<<<gemmGrid, 256, 0, stream>>>(x, W1t, B0, N);
    zero_f32<<<1024, 256, 0, stream>>>(B1, NF);
    edge_scatter<<<edgeGrid, 256, 0, stream>>>(src, dst, ew, dinv, B0, B1, E);
    self_bias<<<1024, 256, 0, stream>>>(B0, dinv, b1, B1, N, /*relu=*/1);

    // layer 2: h_pre = h1@W2 ; aggregate ; +self +b2
    gemm_bf16_wmma<<<gemmGrid, 256, 0, stream>>>(B1, W2t, B0, N);
    zero_f32<<<1024, 256, 0, stream>>>(B1, NF);
    edge_scatter<<<edgeGrid, 256, 0, stream>>>(src, dst, ew, dinv, B0, B1, E);
    self_bias<<<1024, 256, 0, stream>>>(B0, dinv, b2, B1, N, /*relu=*/0);

    // mean pool + heads
    zero_f32<<<1, 128, 0, stream>>>(gsum, 128);
    col_sum<<<512, 128, 0, stream>>>(B1, gsum, N);
    head_kernel<<<1, 128, 0, stream>>>(gsum, Wmu, bmu, Wlv, blv, (float*)d_out, N);
}
